// EEGToLatentGAT_67396626809214
// MI455X (gfx1250) — matrix operand order repaired and checked by reading, not compiled, not run on hardware
//
#include <hip/hip_runtime.h>
#include <hip/hip_bf16.h>
#include <math.h>

typedef __attribute__((ext_vector_type(16))) _Float16 v16h;
typedef __attribute__((ext_vector_type(8)))  _Float16 v8h;
typedef __attribute__((ext_vector_type(8)))  float    v8f;

#define B_   512
#define C_   17
#define T_   80
#define H_   4
#define FG   128    // F_GAT
#define FT   64     // F_TCN
#define KC   384    // conv GEMM K = 3*128 (kk = k*128 + f)
#define LAT  1024
#define HID  256
#define KIN  1088   // C_*FT

#if defined(__has_builtin)
#if __has_builtin(__builtin_amdgcn_sched_barrier)
#define SCHED_BARRIER() __builtin_amdgcn_sched_barrier(0)
#endif
#endif
#ifndef SCHED_BARRIER
#define SCHED_BARRIER()
#endif

// ---------------------------------------------------------------------------
// WMMA fragment loaders (wave32, 16x16x32 f16).
// A (16xK, row-major source, rows = M):
//   lane: m = row0 + (lane&15), half = lane>>4
//   elems j -> K = kbase + 16*half + j           (32B contiguous)
// B (Kx16, N-major source rows of length ld):
//   lane: n = nbase + (lane&15), half = lane>>4
//   elems j -> K = kbase + 16*half + j           (32B contiguous)
// ---------------------------------------------------------------------------
__device__ __forceinline__ v16h load_a16(const _Float16* base, int ld, int row0, int kbase) {
  int lane = threadIdx.x & 31;
  const _Float16* p = base + (row0 + (lane & 15)) * ld + kbase + ((lane >> 4) << 4);
  v8h lo = *(const v8h*)p;
  v8h hi = *(const v8h*)(p + 8);
  v16h a;
#pragma unroll
  for (int j = 0; j < 8; ++j) { a[j] = lo[j]; a[j + 8] = hi[j]; }
  return a;
}

__device__ __forceinline__ v16h load_b16(const _Float16* base, int ld, int nbase, int kbase) {
  int lane = threadIdx.x & 31;
  const _Float16* p = base + (nbase + (lane & 15)) * ld + kbase + ((lane >> 4) << 4);
  v8h lo = *(const v8h*)p;
  v8h hi = *(const v8h*)(p + 8);
  v16h b;
#pragma unroll
  for (int j = 0; j < 8; ++j) { b[j] = lo[j]; b[j + 8] = hi[j]; }
  return b;
}

// ---------------------------------------------------------------------------
// Kernel 1: GAT collapsed to per-head attention-weighted averages.
// m[b,c_i,t,h] = sum_j softmax_j(leaky(x_i*d_h + x_j*s_h, 0.2)) * x_j
// Layout: m_ws[((b*C + i)*T + t)*H + h]  (contiguous 320 floats per (b,i))
// ---------------------------------------------------------------------------
__global__ __launch_bounds__(256) void gat_kernel(
    const float* __restrict__ x, const float* __restrict__ gat_w,
    const float* __restrict__ att_src, const float* __restrict__ att_dst,
    float* __restrict__ m_out) {
  __shared__ float xs[T_ * C_];   // [t][c]
  __shared__ float sh[H_], dh[H_];
  int b = blockIdx.x, tid = threadIdx.x;
  for (int e = tid; e < C_ * T_; e += 256) {
    int c = e / T_, t = e % T_;
    xs[t * C_ + c] = x[(b * C_ + c) * T_ + t];
  }
  if (tid < 8) {
    int h = tid & 3;
    const float* att = (tid < 4) ? att_src : att_dst;
    float s = 0.f;
#pragma unroll
    for (int d = 0; d < 32; ++d) s += gat_w[h * 32 + d] * att[h * 32 + d];
    if (tid < 4) sh[h] = s; else dh[h] = s;
  }
  __syncthreads();
  for (int e = tid; e < T_ * C_ * H_; e += 256) {
    int h = e & 3;
    int ti = e >> 2;
    int i = ti % C_, t = ti / C_;
    const float* xr = &xs[t * C_];
    float xi = xr[i];
    float base = xi * dh[h];
    float s = sh[h];
    float mx = -1e30f;
#pragma unroll
    for (int j = 0; j < C_; ++j) {
      float l = base + xr[j] * s;
      l = l > 0.f ? l : 0.2f * l;
      mx = fmaxf(mx, l);
    }
    float den = 0.f, num = 0.f;
#pragma unroll
    for (int j = 0; j < C_; ++j) {
      float l = base + xr[j] * s;
      l = l > 0.f ? l : 0.2f * l;
      float ee = __expf(l - mx);
      den += ee; num += ee * xr[j];
    }
    m_out[((b * C_ + i) * T_ + t) * H_ + h] = num / den;
  }
}

// ---------------------------------------------------------------------------
// Kernel 2: weight repack/convert to f16.
//  wre[o][k*128+f] = tcn_w[o][f][k]       (64x384)
//  w1t[n][K]       = w1[K][n]             (256x1088, N-major)
//  w2t[n][K]       = w2[K][n]             (1024x256, N-major)
// ---------------------------------------------------------------------------
__global__ void prep_kernel(const float* __restrict__ tcn_w,
                            const float* __restrict__ w1,
                            const float* __restrict__ w2,
                            _Float16* __restrict__ wre,
                            _Float16* __restrict__ w1t,
                            _Float16* __restrict__ w2t) {
  int i = blockIdx.x * blockDim.x + threadIdx.x;
  const int N0 = FT * KC;        // 24576
  const int N1 = HID * KIN;      // 278528
  const int N2 = LAT * HID;      // 262144
  if (i < N0) {
    int o = i / KC, kk = i % KC, k = kk >> 7, f = kk & 127;
    wre[i] = (_Float16)tcn_w[(o * FG + f) * 3 + k];
  } else if (i < N0 + N1) {
    int j = i - N0; int n = j / KIN, K = j % KIN;
    w1t[j] = (_Float16)w1[K * HID + n];
  } else if (i < N0 + N1 + N2) {
    int j = i - N0 - N1; int n = j / HID, K = j % HID;
    w2t[j] = (_Float16)w2[K * LAT + n];
  }
}

// ---------------------------------------------------------------------------
// Kernel 3: fused feats -> conv1d(k=3,pad=1) GEMM -> bias/leaky/BN -> mean_t.
// One block per (b,c): Y[64,80] = Wre[64,384] x X[384,80],
// X[(k*128+f), t] = feats[t+k-1][f] read from zero-padded LDS tile.
// Fully-unrolled K loop; sched_barrier splits each K-step into
// [A global load + 5 B ds-load bursts] | [5 WMMAs], forcing distinct B
// fragment registers so the ds loads pipeline instead of one wait per WMMA.
// ---------------------------------------------------------------------------
__global__ __launch_bounds__(128) void conv_kernel(
    const float* __restrict__ m_in, const float* __restrict__ gat_w,
    const float* __restrict__ gat_bias, const _Float16* __restrict__ wre,
    const float* __restrict__ tcn_b, const float* __restrict__ bn_gamma,
    const float* __restrict__ bn_beta, const float* __restrict__ bn_mean,
    const float* __restrict__ bn_var, _Float16* __restrict__ readout) {
  __shared__ _Float16 feats[(T_ + 2) * FG];   // rows 0 and 81 are zero pad
  __shared__ float ms[T_ * H_];
  __shared__ float sc[FT], sf[FT];
  int bc = blockIdx.x, tid = threadIdx.x;
  int b = bc / C_, c = bc % C_;

  for (int e = tid; e < T_ * H_; e += 128) ms[e] = m_in[bc * (T_ * H_) + e];
  if (tid < FT) {
    float scv = bn_gamma[tid] * rsqrtf(bn_var[tid] + 1e-5f);
    sc[tid] = scv;
    sf[tid] = bn_beta[tid] - bn_mean[tid] * scv;
  }
  __syncthreads();
  for (int e = tid; e < (T_ + 2) * FG; e += 128) {
    int rt = e >> 7, f = e & 127;
    float v = 0.f;
    if (rt > 0 && rt <= T_) {
      float z = ms[(rt - 1) * H_ + (f >> 5)] * gat_w[f] + gat_bias[f];
      v = z > 0.f ? z : (__expf(z) - 1.f);   // elu, alpha=1
    }
    feats[e] = (_Float16)v;
  }
  __syncthreads();

  int wave = tid >> 5, lane = tid & 31;
  int half = lane >> 4, nl = lane & 15;
  int mt = wave;   // o-rows [16*mt, 16*mt+16)

  v8f acc[5];
#pragma unroll
  for (int q = 0; q < 5; ++q)
#pragma unroll
    for (int r = 0; r < 8; ++r) acc[q][r] = 0.f;

#pragma unroll
  for (int kt = 0; kt < KC / 32; ++kt) {
    v16h a = load_a16(wre, KC, mt * 16, kt * 32);
    int kk0 = kt * 32 + half * 16;           // 16-run never crosses a k block
    int kof = kk0 >> 7, f0 = kk0 & 127;
    v16h bb[5];
#pragma unroll
    for (int nt = 0; nt < 5; ++nt) {
      const _Float16* pb = feats + (nt * 16 + nl + kof) * FG + f0;  // +1 pad -1 shift
      v8h blo = *(const v8h*)pb;
      v8h bhi = *(const v8h*)(pb + 8);
#pragma unroll
      for (int j = 0; j < 8; ++j) { bb[nt][j] = blo[j]; bb[nt][j + 8] = bhi[j]; }
    }
    SCHED_BARRIER();   // keep all 5 B fragments live -> clause of ds loads
#pragma unroll
    for (int nt = 0; nt < 5; ++nt)
      acc[nt] = __builtin_amdgcn_wmma_f32_16x16x32_f16(false, a, false, bb[nt],
                                                       (short)0, acc[nt], false, false);
  }

  // Epilogue: +tcn_b, leaky(0.01), BN affine, mean over t, write f16 readout.
#pragma unroll
  for (int r = 0; r < 8; ++r) {
    int o = mt * 16 + r + half * 8;          // C/D layout: M = r + 8*(lane>>4)
    float tb = tcn_b[o], scv = sc[o], sfv = sf[o];
    float s = 0.f;
#pragma unroll
    for (int nt = 0; nt < 5; ++nt) {
      float v = acc[nt][r] + tb;
      v = v > 0.f ? v : 0.01f * v;
      s += v * scv + sfv;
    }
    s += __shfl_xor(s, 1, 32);
    s += __shfl_xor(s, 2, 32);
    s += __shfl_xor(s, 4, 32);
    s += __shfl_xor(s, 8, 32);               // keep halves separate (two o values)
    if (nl == 0) readout[b * KIN + c * FT + o] = (_Float16)(s * (1.f / T_));
  }
}

// ---------------------------------------------------------------------------
// Kernel 4: H1 = leaky(readout @ w1 + b1)  [512,256], K=1088, f16 out.
// Each wave: one M-tile x 2 N-tiles (A fragment shared across 2 WMMAs).
// 32 mt x 8 ngroups = 256 wave-tasks -> 64 blocks of 4 waves.
// ---------------------------------------------------------------------------
__global__ __launch_bounds__(128) void mlp1_kernel(
    const _Float16* __restrict__ ro, const _Float16* __restrict__ w1t,
    const float* __restrict__ b1, _Float16* __restrict__ h1) {
  int wave = threadIdx.x >> 5, lane = threadIdx.x & 31;
  int half = lane >> 4, nl = lane & 15;
  int task = blockIdx.x * 4 + wave;      // 0..255
  int mt = task >> 3, ng = task & 7;     // 32 x 8
  v8f acc[2];
#pragma unroll
  for (int q = 0; q < 2; ++q)
#pragma unroll
    for (int r = 0; r < 8; ++r) acc[q][r] = 0.f;
  for (int kt = 0; kt < KIN / 32; ++kt) {
    v16h a = load_a16(ro, KIN, mt * 16, kt * 32);
    v16h b0 = load_b16(w1t, KIN, (ng * 2 + 0) * 16, kt * 32);
    v16h b1f = load_b16(w1t, KIN, (ng * 2 + 1) * 16, kt * 32);
    SCHED_BARRIER();
    acc[0] = __builtin_amdgcn_wmma_f32_16x16x32_f16(false, a, false, b0,
                                                    (short)0, acc[0], false, false);
    acc[1] = __builtin_amdgcn_wmma_f32_16x16x32_f16(false, a, false, b1f,
                                                    (short)0, acc[1], false, false);
  }
#pragma unroll
  for (int q = 0; q < 2; ++q) {
    int n = (ng * 2 + q) * 16 + nl;
    float bn = b1[n];
#pragma unroll
    for (int r = 0; r < 8; ++r) {
      int m = mt * 16 + r + half * 8;
      float v = acc[q][r] + bn;
      v = v > 0.f ? v : 0.01f * v;
      h1[m * HID + n] = (_Float16)v;
    }
  }
}

// ---------------------------------------------------------------------------
// Kernel 5: out = H1 @ w2 + b2  [512,1024] f32, K=256.
// Each wave: one M-tile x 4 N-tiles (A shared across 4 WMMAs).
// 32 mt x 16 ngroups = 512 wave-tasks -> 128 blocks of 4 waves.
// ---------------------------------------------------------------------------
__global__ __launch_bounds__(128) void mlp2_kernel(
    const _Float16* __restrict__ h1, const _Float16* __restrict__ w2t,
    const float* __restrict__ b2, float* __restrict__ out) {
  int wave = threadIdx.x >> 5, lane = threadIdx.x & 31;
  int half = lane >> 4, nl = lane & 15;
  int task = blockIdx.x * 4 + wave;      // 0..511
  int mt = task >> 4, ng = task & 15;    // 32 x 16
  v8f acc[4];
#pragma unroll
  for (int q = 0; q < 4; ++q)
#pragma unroll
    for (int r = 0; r < 8; ++r) acc[q][r] = 0.f;
  for (int kt = 0; kt < HID / 32; ++kt) {
    v16h a = load_a16(h1, HID, mt * 16, kt * 32);
    v16h bb[4];
#pragma unroll
    for (int q = 0; q < 4; ++q)
      bb[q] = load_b16(w2t, HID, (ng * 4 + q) * 16, kt * 32);
    SCHED_BARRIER();
#pragma unroll
    for (int q = 0; q < 4; ++q)
      acc[q] = __builtin_amdgcn_wmma_f32_16x16x32_f16(false, a, false, bb[q],
                                                      (short)0, acc[q], false, false);
  }
#pragma unroll
  for (int q = 0; q < 4; ++q) {
    int n = (ng * 4 + q) * 16 + nl;
    float bn = b2[n];
#pragma unroll
    for (int r = 0; r < 8; ++r) {
      int m = mt * 16 + r + half * 8;
      out[m * LAT + n] = acc[q][r] + bn;
    }
  }
}

// ---------------------------------------------------------------------------
static inline size_t alup(size_t x) { return (x + 255) & ~(size_t)255; }

extern "C" void kernel_launch(void* const* d_in, const int* in_sizes, int n_in,
                              void* d_out, int out_size, void* d_ws, size_t ws_size,
                              hipStream_t stream) {
  const float* x        = (const float*)d_in[0];
  const float* gat_w    = (const float*)d_in[1];
  const float* att_src  = (const float*)d_in[2];
  const float* att_dst  = (const float*)d_in[3];
  const float* gat_bias = (const float*)d_in[4];
  const float* tcn_w    = (const float*)d_in[5];
  const float* tcn_b    = (const float*)d_in[6];
  const float* bn_gamma = (const float*)d_in[7];
  const float* bn_beta  = (const float*)d_in[8];
  const float* bn_mean  = (const float*)d_in[9];
  const float* bn_var   = (const float*)d_in[10];
  const float* w1       = (const float*)d_in[11];
  const float* b1       = (const float*)d_in[12];
  const float* w2       = (const float*)d_in[13];
  const float* b2       = (const float*)d_in[14];

  char* ws = (char*)d_ws;
  size_t off = 0;
  float*    m_ws   = (float*)(ws + off);    off += alup((size_t)B_ * C_ * T_ * H_ * 4);
  _Float16* wre    = (_Float16*)(ws + off); off += alup((size_t)FT * KC * 2);
  _Float16* w1t    = (_Float16*)(ws + off); off += alup((size_t)HID * KIN * 2);
  _Float16* w2t    = (_Float16*)(ws + off); off += alup((size_t)LAT * HID * 2);
  _Float16* rdo    = (_Float16*)(ws + off); off += alup((size_t)B_ * KIN * 2);
  _Float16* h1     = (_Float16*)(ws + off); off += alup((size_t)B_ * HID * 2);

  gat_kernel<<<B_, 256, 0, stream>>>(x, gat_w, att_src, att_dst, m_ws);

  int prep_total = FT * KC + HID * KIN + LAT * HID;
  prep_kernel<<<(prep_total + 255) / 256, 256, 0, stream>>>(tcn_w, w1, w2, wre, w1t, w2t);

  conv_kernel<<<B_ * C_, 128, 0, stream>>>(m_ws, gat_w, gat_bias, wre, tcn_b,
                                           bn_gamma, bn_beta, bn_mean, bn_var, rdo);

  mlp1_kernel<<<64, 128, 0, stream>>>(rdo, w1t, b1, h1);

  mlp2_kernel<<<128, 128, 0, stream>>>(h1, w2t, b2, (float*)d_out);
}
